// GPT2Attention_17918603559271
// MI455X (gfx1250) — compile-verified, matrix operand order checked by reading, not run on hardware
//
#include <hip/hip_runtime.h>
#include <hip/hip_bf16.h>

// ---------------------------------------------------------------------------
// Types for CDNA5 WMMA (wave32): 16x16x32 bf16 -> f32
// ---------------------------------------------------------------------------
typedef __attribute__((ext_vector_type(16))) __bf16 v16bf;
typedef __attribute__((ext_vector_type(8)))  float  v8f;
typedef __attribute__((ext_vector_type(4)))  unsigned int v4u;
typedef __attribute__((ext_vector_type(4)))  float  v4f;
typedef __attribute__((ext_vector_type(4)))  __bf16 v4bf;

#define WMMA_BF16(A, B, C) \
  __builtin_amdgcn_wmma_f32_16x16x32_bf16(false, (A), false, (B), (short)0, (C), false, false)

static constexpr int   kB = 4, kS = 2048, kH = 1024, kHeads = 16, kDh = 64;
static constexpr float kLog2Base = 13.287712379549449f; // log2(10000)

union FragU { v16bf v; v4u q[2]; };

// Load one 16x32 bf16 A/B fragment. Per the CDNA5 16-bit operand layout:
//   lanes 0-15 : row = lane,    K chunks {k0+0..7, k0+16..23}
//   lanes 16-31: row = lane-16, K chunks {k0+8..15, k0+24..31}
// Both chunks are 16B-contiguous -> two b128 global loads per lane.
__device__ __forceinline__ v16bf load_frag(const __bf16* __restrict__ base,
                                           int stride, int row, int k0, int lane) {
  const int l15  = lane & 15;
  const int koff = k0 + ((lane >> 4) << 3);
  const __bf16* p = base + (size_t)(row + l15) * stride + koff;
  FragU u;
  u.q[0] = *(const v4u*)(p);
  u.q[1] = *(const v4u*)(p + 16);
  return u.v;
}

// ---------------------------------------------------------------------------
// fp32 -> bf16 conversion (b128 in, b64 out)
// ---------------------------------------------------------------------------
__global__ __launch_bounds__(256) void cvt_bf16(const float* __restrict__ src,
                                                __bf16* __restrict__ dst, int n) {
  int i = (blockIdx.x * 256 + threadIdx.x) * 4;
  if (i >= n) return;
  v4f v = *(const v4f*)(src + i);
  v4bf o;
  o.x = (__bf16)v.x; o.y = (__bf16)v.y; o.z = (__bf16)v.z; o.w = (__bf16)v.w;
  *(v4bf*)(dst + i) = o;
}

// ---------------------------------------------------------------------------
// Projection GEMM: Y[m,n] = sum_k X[m,k] * W[n,k] + bias[n]
//   mode 0: Q  -> RoPE + 1/sqrt(dh) scale, store bf16 [b,h,S,64]
//   mode 1: K  -> RoPE,                    store bf16 [b,h,S,64]
//   mode 2: V  ->                          store bf16 [b,h,64,S]  (transposed)
//   mode 3: O-proj ->                      store f32  [m,1024]
// Wave tile: 32x64 (2 M-frags x 4 N-frags, 8 accumulators). Block = 4 waves.
// N tile (64) == one head, so RoPE's d<->d+32 pairing stays in-lane across
// accumulator registers (acc ni <-> ni+2).
// ---------------------------------------------------------------------------
__global__ __launch_bounds__(128) void proj_gemm(
    const __bf16* __restrict__ X, const __bf16* __restrict__ W,
    const float* __restrict__ bias, __bf16* __restrict__ dst_bf,
    float* __restrict__ dst_f32, int mode)
{
  const int lane  = threadIdx.x & 31;
  const int wave  = threadIdx.x >> 5;
  const int l15   = lane & 15;
  const int half8 = (lane >> 4) << 3;
  const int mwave = blockIdx.x * 128 + wave * 32;
  const int nbase = blockIdx.y * 64;
  const int head  = nbase >> 6;

  const v8f vzero = {0.f, 0.f, 0.f, 0.f, 0.f, 0.f, 0.f, 0.f};
  v8f acc[2][4];
#pragma unroll
  for (int mi = 0; mi < 2; ++mi)
#pragma unroll
    for (int ni = 0; ni < 4; ++ni) acc[mi][ni] = vzero;

  for (int k0 = 0; k0 < kH; k0 += 32) {
    v16bf a[2], b[4];
#pragma unroll
    for (int mi = 0; mi < 2; ++mi) a[mi] = load_frag(X, kH, mwave + mi * 16, k0, lane);
#pragma unroll
    for (int ni = 0; ni < 4; ++ni) b[ni] = load_frag(W, kH, nbase + ni * 16, k0, lane);
#pragma unroll
    for (int mi = 0; mi < 2; ++mi)
#pragma unroll
      for (int ni = 0; ni < 4; ++ni)
        acc[mi][ni] = WMMA_BF16(a[mi], b[ni], acc[mi][ni]);
  }

  float bv[4];
#pragma unroll
  for (int ni = 0; ni < 4; ++ni) bv[ni] = bias[nbase + ni * 16 + l15];

  // RoPE inverse frequencies for this lane's two column parities (d%32).
  const float invf0 = exp2f(-(float)(l15)      * (kLog2Base / 32.f));
  const float invf1 = exp2f(-(float)(16 + l15) * (kLog2Base / 32.f));

#pragma unroll
  for (int mi = 0; mi < 2; ++mi) {
#pragma unroll
    for (int r = 0; r < 8; ++r) {
      const int m = mwave + mi * 16 + r + half8;   // global row
      const int bb = m >> 11;                      // batch
      const int s  = m & 2047;                     // seq position

      if (mode <= 1) {                             // Q or K: bias + RoPE
        float sn0, cs0, sn1, cs1;
        const float fs = (float)s;
        __sincosf(fs * invf0, &sn0, &cs0);
        __sincosf(fs * invf1, &sn1, &cs1);
        float y0 = acc[mi][0][r] + bv[0];
        float y1 = acc[mi][1][r] + bv[1];
        float y2 = acc[mi][2][r] + bv[2];
        float y3 = acc[mi][3][r] + bv[3];
        float o0 = y0 * cs0 - y2 * sn0;
        float o1 = y1 * cs1 - y3 * sn1;
        float o2 = y2 * cs0 + y0 * sn0;
        float o3 = y3 * cs1 + y1 * sn1;
        if (mode == 0) { o0 *= 0.125f; o1 *= 0.125f; o2 *= 0.125f; o3 *= 0.125f; }
        const size_t rowb = (((size_t)((bb << 4) + head) << 11) + (size_t)s) << 6;
        dst_bf[rowb +  0 + l15] = (__bf16)o0;
        dst_bf[rowb + 16 + l15] = (__bf16)o1;
        dst_bf[rowb + 32 + l15] = (__bf16)o2;
        dst_bf[rowb + 48 + l15] = (__bf16)o3;
      } else if (mode == 2) {                      // V: store transposed [bh,64,S]
#pragma unroll
        for (int ni = 0; ni < 4; ++ni) {
          const int d = ni * 16 + l15;
          const size_t addr = (((size_t)((bb << 4) + head) * 64 + d) << 11) + (size_t)s;
          dst_bf[addr] = (__bf16)(acc[mi][ni][r] + bv[ni]);
        }
      } else {                                     // O-proj: f32 output
#pragma unroll
        for (int ni = 0; ni < 4; ++ni)
          dst_f32[(size_t)m * kH + nbase + ni * 16 + l15] = acc[mi][ni][r] + bv[ni];
      }
    }
  }
}

// ---------------------------------------------------------------------------
// Flash attention (per wave: 16 queries x full 64-dim, streaming 64-key blocks)
// Computes S^T = K·Q^T so the softmax key-reduction is per-lane; the P^T f32
// accumulators repack in-lane into bf16 B fragments for O^T += V^T·P^T.
// ---------------------------------------------------------------------------
__global__ __launch_bounds__(128) void flash_attn(
    const __bf16* __restrict__ Q, const __bf16* __restrict__ K,
    const __bf16* __restrict__ Vt, __bf16* __restrict__ Attn)
{
  const int lane  = threadIdx.x & 31;
  const int wave  = threadIdx.x >> 5;
  const int l15   = lane & 15;
  const int half8 = (lane >> 4) << 3;
  const int bh    = blockIdx.y;                 // b*16 + head
  const int qbase = blockIdx.x * 64 + wave * 16;

  const __bf16* Qh = Q  + (size_t)bh * (kS * kDh);
  const __bf16* Kh = K  + (size_t)bh * (kS * kDh);
  const __bf16* Vh = Vt + (size_t)bh * (kDh * kS);

  // Q^T B-fragments (d = K-dim), resident for the whole wave.
  const v16bf qf0 = load_frag(Qh, kDh, qbase, 0,  lane);
  const v16bf qf1 = load_frag(Qh, kDh, qbase, 32, lane);

  const v8f vzero = {0.f, 0.f, 0.f, 0.f, 0.f, 0.f, 0.f, 0.f};
  v8f accO[4];
#pragma unroll
  for (int mt = 0; mt < 4; ++mt) accO[mt] = vzero;
  float mrun = -3.0e38f, lrun = 0.f;

  for (int kb = 0; kb < kS; kb += 64) {
    // S^T[key, q] for 64 keys (4 accumulators of 16 keys each)
    v8f s4[4];
#pragma unroll
    for (int kt = 0; kt < 4; ++kt) {
      v16bf ka0 = load_frag(Kh, kDh, kb + kt * 16, 0,  lane);
      v16bf ka1 = load_frag(Kh, kDh, kb + kt * 16, 32, lane);
      v8f z = vzero;
      z = WMMA_BF16(ka0, qf0, z);
      z = WMMA_BF16(ka1, qf1, z);
      s4[kt] = z;
    }

    // Online softmax stats (per-lane over 32 keys + one xor-16 merge).
    float mloc = mrun;
#pragma unroll
    for (int kt = 0; kt < 4; ++kt)
#pragma unroll
      for (int r = 0; r < 8; ++r) mloc = fmaxf(mloc, s4[kt][r]);
    mloc = fmaxf(mloc, __shfl_xor(mloc, 16, 32));
    const float alpha = __expf(mrun - mloc);
    mrun = mloc;

    // exp() + in-lane repack of P^T accumulators into bf16 B fragments.
    float lsum = 0.f;
    v16bf pf0, pf1;
#pragma unroll
    for (int h = 0; h < 16; ++h) {
      const float e0 = __expf(s4[h >> 3][h & 7] - mloc);
      const float e1 = __expf(s4[2 + (h >> 3)][h & 7] - mloc);
      lsum += e0 + e1;
      pf0[h] = (__bf16)e0;
      pf1[h] = (__bf16)e1;
    }
    lsum += __shfl_xor(lsum, 16, 32);
    lrun = lrun * alpha + lsum;

    // O^T += V^T · P^T (4 d-tiles x 2 key halves = 8 WMMAs)
#pragma unroll
    for (int mt = 0; mt < 4; ++mt) {
#pragma unroll
      for (int r = 0; r < 8; ++r) accO[mt][r] *= alpha;
      v16bf va0 = load_frag(Vh, kS, mt * 16, kb,      lane);
      v16bf va1 = load_frag(Vh, kS, mt * 16, kb + 32, lane);
      accO[mt] = WMMA_BF16(va0, pf0, accO[mt]);
      accO[mt] = WMMA_BF16(va1, pf1, accO[mt]);
    }
  }

  const float inv = 1.f / lrun;
  const int q = qbase + l15;
  const int bb = bh >> 4, head = bh & 15;
  const size_t rowb = ((size_t)(bb * kS + q)) * kH + head * kDh;
#pragma unroll
  for (int mt = 0; mt < 4; ++mt)
#pragma unroll
    for (int r = 0; r < 8; ++r) {
      const int d = mt * 16 + r + half8;
      Attn[rowb + d] = (__bf16)(accO[mt][r] * inv);
    }
}

// ---------------------------------------------------------------------------
// Host launch
// ---------------------------------------------------------------------------
extern "C" void kernel_launch(void* const* d_in, const int* in_sizes, int n_in,
                              void* d_out, int out_size, void* d_ws, size_t ws_size,
                              hipStream_t stream) {
  (void)in_sizes; (void)n_in; (void)out_size; (void)ws_size;
  const float* hidden = (const float*)d_in[0];
  const float* Wq = (const float*)d_in[1];
  const float* bq = (const float*)d_in[2];
  const float* Wk = (const float*)d_in[3];
  const float* bk = (const float*)d_in[4];
  const float* Wv = (const float*)d_in[5];
  const float* bv = (const float*)d_in[6];
  const float* Wo = (const float*)d_in[7];
  const float* bo = (const float*)d_in[8];

  char* ws = (char*)d_ws;                                   // workspace map (MB):
  __bf16* Xb  = (__bf16*)(ws);                              // [ 0,16) X bf16
  __bf16* Wqb = (__bf16*)(ws + ((size_t)16 << 20));         // [16,18)
  __bf16* Wkb = (__bf16*)(ws + ((size_t)18 << 20));         // [18,20)
  __bf16* Wvb = (__bf16*)(ws + ((size_t)20 << 20));         // [20,22)
  __bf16* Wob = (__bf16*)(ws + ((size_t)22 << 20));         // [22,24)
  __bf16* Qb  = (__bf16*)(ws + ((size_t)24 << 20));         // [24,40) RoPE'd+scaled Q
  __bf16* Kb  = (__bf16*)(ws + ((size_t)40 << 20));         // [40,56) RoPE'd K
  __bf16* Vtb = (__bf16*)(ws + ((size_t)56 << 20));         // [56,72) V transposed
  __bf16* Ab  = (__bf16*)(ws + ((size_t)72 << 20));         // [72,88) attn out bf16

  const int nX = kB * kS * kH;      // 8388608
  const int nW = kH * kH;           // 1048576
  cvt_bf16<<<nX / 1024, 256, 0, stream>>>(hidden, Xb, nX);
  cvt_bf16<<<nW / 1024, 256, 0, stream>>>(Wq, Wqb, nW);
  cvt_bf16<<<nW / 1024, 256, 0, stream>>>(Wk, Wkb, nW);
  cvt_bf16<<<nW / 1024, 256, 0, stream>>>(Wv, Wvb, nW);
  cvt_bf16<<<nW / 1024, 256, 0, stream>>>(Wo, Wob, nW);

  dim3 gGemm(64, 16), blk(128);
  proj_gemm<<<gGemm, blk, 0, stream>>>(Xb, Wqb, bq, Qb,  nullptr, 0);
  proj_gemm<<<gGemm, blk, 0, stream>>>(Xb, Wkb, bk, Kb,  nullptr, 1);
  proj_gemm<<<gGemm, blk, 0, stream>>>(Xb, Wvb, bv, Vtb, nullptr, 2);

  flash_attn<<<dim3(32, 64), blk, 0, stream>>>(Qb, Kb, Vtb, Ab);

  proj_gemm<<<gGemm, blk, 0, stream>>>(Ab, Wob, bo, nullptr, (float*)d_out, 3);
}
